// MaskedCrossAttention_41618233098501
// MI455X (gfx1250) — compile-verified, matrix operand order checked by reading, not run on hardware
//
#include <hip/hip_runtime.h>
#include <hip/hip_bf16.h>
#include <stdint.h>

// ---------------- problem constants (from reference) ----------------
constexpr int BB     = 4;
constexpr int TT     = 2048;
constexpr int DIMC   = 2048;
constexpr int DIMV   = 1024;
constexpr int INNERC = 1024;   // HEADS * DIM_HEAD
constexpr int HEADSC = 16;
constexpr int DHEAD  = 64;
constexpr int JTOT   = 512;    // N_MEDIA * N_VISUAL keys per batch
constexpr float LNEPS = 1e-5f;
constexpr float QSCALE = 0.125f; // 64^-0.5

// ---------------- CDNA5 WMMA types ----------------
typedef __bf16 bf16;
typedef __attribute__((ext_vector_type(16))) __bf16 v16bf;
typedef __attribute__((ext_vector_type(8)))  __bf16 v8bf;
typedef __attribute__((ext_vector_type(8)))  float  v8f;
typedef __attribute__((ext_vector_type(4)))  int    v4i;

#define GLOBAL_AS __attribute__((address_space(1)))
#define LDS_AS    __attribute__((address_space(3)))

// CDNA5 async global->LDS path (ASYNCcnt), guarded so we fall back cleanly if
// this toolchain doesn't declare the builtins. Probe-verified on this toolchain:
// param0 = 128-bit int vector pointer (global AS), param1 = same in LDS AS.
#if defined(__has_builtin)
#if __has_builtin(__builtin_amdgcn_global_load_async_to_lds_b128) && \
    __has_builtin(__builtin_amdgcn_s_wait_asynccnt)
#define USE_ASYNC_LDS 1
#endif
#endif
#ifndef USE_ASYNC_LDS
#define USE_ASYNC_LDS 0
#endif

static __device__ __forceinline__ bf16 f2bf(float f) {
  // round-to-nearest-even fp32 -> bf16
  unsigned u = __builtin_bit_cast(unsigned, f);
  unsigned r = (u + 0x7FFFu + ((u >> 16) & 1u)) >> 16;
  unsigned short h = (unsigned short)r;
  return __builtin_bit_cast(bf16, h);
}

// ---------------- kernel 1: per-batch cumsum of media locations ----------------
__global__ void mca_cumsum_kernel(const unsigned char* __restrict__ ml,
                                  int* __restrict__ tt) {
  if (threadIdx.x != 0) return;
  const int b = blockIdx.x;
  int s = 0;
  for (int t = 0; t < TT; ++t) {
    s += (ml[b * TT + t] != 0);
    tt[b * TT + t] = s;
  }
}

// ---------------- kernel 2: LayerNorm (fp32 in) -> bf16 out ----------------
__global__ __launch_bounds__(256)
void mca_ln_kernel(const float* __restrict__ y, const float* __restrict__ g,
                   const float* __restrict__ be, bf16* __restrict__ yn) {
  __shared__ float redS[8], redS2[8];
  const int row = blockIdx.x;          // 0 .. B*T-1
  const int tid = threadIdx.x;
  const float* x = y + (size_t)row * DIMC;
  float v[8];
  float s = 0.f, s2 = 0.f;
#pragma unroll
  for (int e = 0; e < 8; ++e) {
    v[e] = x[tid + e * 256];
    s += v[e];
    s2 += v[e] * v[e];
  }
#pragma unroll
  for (int msk = 1; msk < 32; msk <<= 1) {
    s  += __shfl_xor(s,  msk, 32);
    s2 += __shfl_xor(s2, msk, 32);
  }
  if ((tid & 31) == 0) { redS[tid >> 5] = s; redS2[tid >> 5] = s2; }
  __syncthreads();
  float ts = 0.f, ts2 = 0.f;
#pragma unroll
  for (int w = 0; w < 8; ++w) { ts += redS[w]; ts2 += redS2[w]; }
  const float mu  = ts * (1.f / DIMC);
  const float var = ts2 * (1.f / DIMC) - mu * mu;
  const float inv = rsqrtf(var + LNEPS);
  bf16* o = yn + (size_t)row * DIMC;
#pragma unroll
  for (int e = 0; e < 8; ++e) {
    const int c = tid + e * 256;
    o[c] = f2bf((v[e] - mu) * inv * g[c] + be[c]);
  }
}

// ---------------- kernel 3: fp32 -> bf16 vector convert ----------------
__global__ __launch_bounds__(256)
void mca_f2bf_kernel(const float* __restrict__ s, bf16* __restrict__ d, int n4) {
  const int i = blockIdx.x * 256 + threadIdx.x;
  if (i >= n4) return;
  const float4 v = ((const float4*)s)[i];
  bf16* o = d + (size_t)i * 4;
  o[0] = f2bf(v.x); o[1] = f2bf(v.y); o[2] = f2bf(v.z); o[3] = f2bf(v.w);
}

// ---------------- kernel 4: tiled WMMA GEMM  C[M,N] = scale * A[M,K] @ W[N,K]^T ----
// block: 256 threads = 8 waves; tile 128(M) x 128(N); K step 32; double-buffered LDS.
// wave (wr 0..3, wc 0..1) owns a 32x64 subtile = 2x4 v_wmma_f32_16x16x32_bf16 accs.
// LDS tile fills go through GLOBAL_LOAD_ASYNC_TO_LDS_B128 (ASYNCcnt) when available:
// the fill for K-tile i+1 is issued before computing tile i, then s_wait_asynccnt(4)
// retires exactly the current buffer's 4 per-thread async ops (in-order completion).
template <bool OUTF32>
__global__ __launch_bounds__(256)
void mca_gemm_kernel(const bf16* __restrict__ A, const bf16* __restrict__ W,
                     void* __restrict__ Cout, int M, int N, int K, float scale) {
  __shared__ __align__(32) bf16 As[2][128][32];
  __shared__ __align__(32) bf16 Bs[2][128][32];
  const int tid  = threadIdx.x;
  const int lane = tid & 31;
  const int wid  = tid >> 5;
  const int wr   = wid >> 1;          // 0..3  -> 32 rows
  const int wc   = wid & 1;           // 0..1  -> 64 cols
  const int l16  = lane & 15;
  const int kh   = (lane >> 4) * 16;  // K half-select: contiguous k = 16*half + s, same for A and B
  const int blockM = blockIdx.y * 128;
  const int blockN = blockIdx.x * 128;

  v8f acc[2][4] = {};

  // each thread moves 2x16B of A and 2x16B of B per K-tile (4 async ops)
  auto fill_tile = [&](int buf, int k0) {
#pragma unroll
    for (int o = 0; o < 2; ++o) {
      const int ia  = tid * 2 + o;          // 0..511
      const int r   = ia >> 2;              // 0..127
      const int sg  = (ia & 3) * 8;         // 0,8,16,24 halfs (16B)
      const bf16* gA = A + (size_t)(blockM + r) * K + k0 + sg;
      const bf16* gB = W + (size_t)(blockN + r) * K + k0 + sg;
      bf16* lA = &As[buf][r][sg];
      bf16* lB = &Bs[buf][r][sg];
#if USE_ASYNC_LDS
      __builtin_amdgcn_global_load_async_to_lds_b128(
          (GLOBAL_AS v4i*)gA, (LDS_AS v4i*)lA, 0, 0);
      __builtin_amdgcn_global_load_async_to_lds_b128(
          (GLOBAL_AS v4i*)gB, (LDS_AS v4i*)lB, 0, 0);
#else
      *(v8bf*)lA = *(const v8bf*)gA;
      *(v8bf*)lB = *(const v8bf*)gB;
#endif
    }
  };

  const int ntiles = K >> 5;
  fill_tile(0, 0);
  for (int ti = 0; ti < ntiles; ++ti) {
    const int cur = ti & 1;
    if (ti + 1 < ntiles) {
      fill_tile(cur ^ 1, (ti + 1) << 5);
#if USE_ASYNC_LDS
      __builtin_amdgcn_s_wait_asynccnt(4);   // retire current buffer's fills
#endif
    } else {
#if USE_ASYNC_LDS
      __builtin_amdgcn_s_wait_asynccnt(0);
#endif
    }
    __syncthreads();

    v16bf af[2], bfr[4];
#pragma unroll
    for (int i = 0; i < 2; ++i)
      af[i] = *(const v16bf*)&As[cur][wr * 32 + i * 16 + l16][kh];
#pragma unroll
    for (int j = 0; j < 4; ++j)
      bfr[j] = *(const v16bf*)&Bs[cur][wc * 64 + j * 16 + l16][kh];

#pragma unroll
    for (int i = 0; i < 2; ++i)
#pragma unroll
      for (int j = 0; j < 4; ++j)
        acc[i][j] = __builtin_amdgcn_wmma_f32_16x16x32_bf16(
            false, af[i], false, bfr[j], (short)0, acc[i][j], false, false);
    __syncthreads();
  }

  // epilogue: C layout -> element (vgpr p, lane): row = p + 8*(lane>=16), col = lane%16
  const int rhalf = 8 * (lane >> 4);
#pragma unroll
  for (int i = 0; i < 2; ++i)
#pragma unroll
    for (int j = 0; j < 4; ++j)
#pragma unroll
      for (int p = 0; p < 8; ++p) {
        const int row = blockM + wr * 32 + i * 16 + p + rhalf;
        const int col = blockN + wc * 64 + j * 16 + l16;
        const float v = acc[i][j][p] * scale;
        if (OUTF32) ((float*)Cout)[(size_t)row * N + col] = v;
        else        ((bf16*) Cout)[(size_t)row * N + col] = f2bf(v);
      }
}

// ---------------- kernel 5: masked cross-attention core ----------------
// one wave per (16-query tile, head). Images every 256 tokens => all 16 queries in
// a 16-aligned tile share the same media block; mask selects exactly 64 keys.
__global__ __launch_bounds__(32)
void mca_attn_kernel(const bf16* __restrict__ q, const bf16* __restrict__ kv,
                     const int* __restrict__ tt, bf16* __restrict__ ao) {
  __shared__ __align__(32) bf16 pLds[16][64];   // normalized probs, A-layout staging
  __shared__ __align__(32) bf16 vT[64][64];     // V transposed: vT[d][j]
  const int lane = threadIdx.x;
  const int l16  = lane & 15;
  const int half = lane >> 4;
  const int kh   = half * 16;
  const int tileId = blockIdx.x;                // 0 .. B*T/16-1
  const int h = blockIdx.y;
  const int b    = tileId / (TT / 16);
  const int trow = (tileId % (TT / 16)) * 16;
  const int t = tt[b * TT + trow];              // uniform across the wave

  if (t == 0) {                                 // no preceding media: zero output
#pragma unroll
    for (int it = 0; it < 32; ++it) {
      const int idx = it * 32 + lane;
      const int r = idx >> 6, d = idx & 63;
      ao[((size_t)(b * TT + trow + r)) * INNERC + h * DHEAD + d] = f2bf(0.f);
    }
    return;
  }
  const int jbase = (t - 1) * 64;               // attended media block

  // stage V^T into LDS (v stored row-major [j][2*INNER], second half is V)
#pragma unroll
  for (int it = 0; it < 16; ++it) {
    const int idx = it * 32 + lane;             // 0..511
    const int j = idx >> 3;
    const int dseg = (idx & 7) * 8;
    const v8bf vv = *(const v8bf*)(kv + ((size_t)(b * JTOT + jbase + j)) * (2 * INNERC)
                                   + INNERC + h * DHEAD + dseg);
#pragma unroll
    for (int e = 0; e < 8; ++e) vT[dseg + e][j] = vv[e];
  }

  // Q fragments: A-matrix 16x32 bf16, two K-chunks covering d=0..63
  v16bf aq[2];
#pragma unroll
  for (int c = 0; c < 2; ++c)
    aq[c] = *(const v16bf*)(q + ((size_t)(b * TT + trow + l16)) * INNERC
                            + h * DHEAD + c * 32 + kh);

  // S = Q @ K^T : 4 key-groups of 16, 2 K-chunks each
  v8f s[4] = {};
#pragma unroll
  for (int n = 0; n < 4; ++n)
#pragma unroll
    for (int c = 0; c < 2; ++c) {
      const v16bf bk = *(const v16bf*)(kv + ((size_t)(b * JTOT + jbase + n * 16 + l16)) * (2 * INNERC)
                                       + h * DHEAD + c * 32 + kh);
      s[n] = __builtin_amdgcn_wmma_f32_16x16x32_bf16(
          false, aq[c], false, bk, (short)0, s[n], false, false);
    }

  // softmax over the 64 keys. C layout: row = p + 8*half, col = n*16 + (lane%16).
  // lanes within a 16-lane half hold the same rows -> reduce with shfl_xor masks 1..8.
  float inv_s[8];
#pragma unroll
  for (int p = 0; p < 8; ++p) {
    float m = fmaxf(fmaxf(s[0][p], s[1][p]), fmaxf(s[2][p], s[3][p]));
#pragma unroll
    for (int msk = 1; msk < 16; msk <<= 1) m = fmaxf(m, __shfl_xor(m, msk, 32));
    float sum = 0.f;
#pragma unroll
    for (int n = 0; n < 4; ++n) {
      const float e = __expf(s[n][p] - m);
      s[n][p] = e;
      sum += e;
    }
#pragma unroll
    for (int msk = 1; msk < 16; msk <<= 1) sum += __shfl_xor(sum, msk, 32);
    inv_s[p] = 1.f / sum;
  }
#pragma unroll
  for (int n = 0; n < 4; ++n)
#pragma unroll
    for (int p = 0; p < 8; ++p)
      pLds[p + 8 * half][n * 16 + l16] = f2bf(s[n][p] * inv_s[p]);
  __syncthreads();

  // O = P @ V : A = probs (16x64, 2 K-chunks), B = vT rows (d fixed per lane col)
  v16bf ap[2];
#pragma unroll
  for (int c = 0; c < 2; ++c) ap[c] = *(const v16bf*)&pLds[l16][c * 32 + kh];
  v8f o[4] = {};
#pragma unroll
  for (int dt = 0; dt < 4; ++dt)
#pragma unroll
    for (int c = 0; c < 2; ++c) {
      const v16bf bv = *(const v16bf*)&vT[dt * 16 + l16][c * 32 + kh];
      o[dt] = __builtin_amdgcn_wmma_f32_16x16x32_bf16(
          false, ap[c], false, bv, (short)0, o[dt], false, false);
    }

#pragma unroll
  for (int dt = 0; dt < 4; ++dt)
#pragma unroll
    for (int p = 0; p < 8; ++p) {
      const int row = trow + p + 8 * half;
      const int d = dt * 16 + l16;
      ao[((size_t)(b * TT + row)) * INNERC + h * DHEAD + d] = f2bf(o[dt][p]);
    }
}

// ---------------- host-side orchestration ----------------
extern "C" void kernel_launch(void* const* d_in, const int* in_sizes, int n_in,
                              void* d_out, int out_size, void* d_ws, size_t ws_size,
                              hipStream_t stream) {
  (void)in_sizes; (void)n_in; (void)out_size; (void)ws_size;
  const float*         y    = (const float*)d_in[0];
  const unsigned char* ml   = (const unsigned char*)d_in[1];  // jnp bool -> 1 byte
  const float*         vf   = (const float*)d_in[2];
  const float*         ln_g = (const float*)d_in[3];
  const float*         ln_b = (const float*)d_in[4];
  const float*         Wq   = (const float*)d_in[5];
  const float*         Wkv  = (const float*)d_in[6];
  const float*         Wo   = (const float*)d_in[7];
  float* out = (float*)d_out;

  // workspace carve-up (256B aligned slabs)
  char* ws = (char*)d_ws;
  size_t off = 0;
  auto carve = [&](size_t bytes) {
    void* p = ws + off;
    off += (bytes + 255) & ~(size_t)255;
    return p;
  };
  int*  tt   = (int*) carve((size_t)BB * TT * 4);
  bf16* yn   = (bf16*)carve((size_t)BB * TT * DIMC * 2);
  bf16* vfb  = (bf16*)carve((size_t)BB * JTOT * DIMV * 2);
  bf16* wqb  = (bf16*)carve((size_t)INNERC * DIMC * 2);
  bf16* wkvb = (bf16*)carve((size_t)2 * INNERC * DIMV * 2);
  bf16* wob  = (bf16*)carve((size_t)DIMC * INNERC * 2);
  bf16* qb   = (bf16*)carve((size_t)BB * TT * INNERC * 2);
  bf16* kvb  = (bf16*)carve((size_t)BB * JTOT * 2 * INNERC * 2);
  bf16* aob  = (bf16*)carve((size_t)BB * TT * INNERC * 2);

  // 1) text_time cumsum
  mca_cumsum_kernel<<<BB, 32, 0, stream>>>(ml, tt);

  // 2) LayerNorm -> bf16
  mca_ln_kernel<<<BB * TT, 256, 0, stream>>>(y, ln_g, ln_b, yn);

  // 3) fp32 -> bf16 conversions
  {
    const int n_vf  = BB * JTOT * DIMV / 4;
    const int n_wq  = INNERC * DIMC / 4;
    const int n_wkv = 2 * INNERC * DIMV / 4;
    const int n_wo  = DIMC * INNERC / 4;
    mca_f2bf_kernel<<<(n_vf  + 255) / 256, 256, 0, stream>>>(vf,  vfb,  n_vf);
    mca_f2bf_kernel<<<(n_wq  + 255) / 256, 256, 0, stream>>>(Wq,  wqb,  n_wq);
    mca_f2bf_kernel<<<(n_wkv + 255) / 256, 256, 0, stream>>>(Wkv, wkvb, n_wkv);
    mca_f2bf_kernel<<<(n_wo  + 255) / 256, 256, 0, stream>>>(Wo,  wob,  n_wo);
  }

  // 4) Q = yn @ Wq^T * scale   [M=8192, N=1024, K=2048]
  mca_gemm_kernel<false><<<dim3(INNERC / 128, (BB * TT) / 128), 256, 0, stream>>>(
      yn, wqb, (void*)qb, BB * TT, INNERC, DIMC, QSCALE);

  // 5) KV = vf @ Wkv^T         [M=2048, N=2048, K=1024]
  mca_gemm_kernel<false><<<dim3((2 * INNERC) / 128, (BB * JTOT) / 128), 256, 0, stream>>>(
      vfb, wkvb, (void*)kvb, BB * JTOT, 2 * INNERC, DIMV, 1.0f);

  // 6) masked cross-attention core (one wave per 16-query tile per head)
  mca_attn_kernel<<<dim3((BB * TT) / 16, HEADSC), 32, 0, stream>>>(qb, kvb, tt, aob);

  // 7) out = attn @ Wo^T (fp32 output)  [M=8192, N=2048, K=1024]
  mca_gemm_kernel<true><<<dim3(DIMC / 128, (BB * TT) / 128), 256, 0, stream>>>(
      aob, wob, (void*)out, BB * TT, DIMC, INNERC, 1.0f);
}
